// my_DDQN_22376779612772
// MI455X (gfx1250) — matrix operand order repaired and checked by reading, not compiled, not run on hardware
//
#include <hip/hip_runtime.h>

typedef __attribute__((ext_vector_type(16))) _Float16 v16h;
typedef __attribute__((ext_vector_type(8)))  float    v8f;
typedef __attribute__((ext_vector_type(2)))  float    v2f;

// M[j][k] for the 69x42 window-mask matrix, computed analytically
// (j in [57,69) rows are zero, reproducing the source bug).
__device__ __forceinline__ float mask_val(int j, int r, int c) {
    if (j < 24) {                       // row windows
        int row = j >> 2, i = j & 3;
        return (r == row && c >= i && c < i + 4) ? 1.0f : 0.0f;
    } else if (j < 45) {                // column windows
        int jj = j - 24;
        int col = jj / 3, i = jj % 3;
        return (c == col && r >= i && r < i + 4) ? 1.0f : 0.0f;
    } else if (j < 57) {                // 'upward diag' = full 4x4 blocks
        int jj = j - 45;
        int off = jj >> 2, col = jj & 3;
        return (r >= 2 - off && r < 6 - off && c >= col && c < col + 4) ? 1.0f : 0.0f;
    }
    return 0.0f;                        // never-written rows
}

// Fold W[:, :42] + W[:, 42:111] @ M into f16 "Weff", stored directly in the
// per-lane WMMA B-operand layout:
//   bop[(w*32 + lane)*16 + h] = Weff[n=lane%16][ K=32*w + (h<8 ? 0 : 16) + (lane<16?0:8) + h%8 ]
// with zeros for n>=7 or K>=42.  (1024 halves = 2 KB in d_ws.)
__global__ void fold_weights(const float* __restrict__ W, _Float16* __restrict__ bop) {
    int tid  = threadIdx.x;            // 0..1023, exactly one half each
    int w    = tid >> 9;               // which WMMA (K block of 32)
    int lane = (tid >> 4) & 31;
    int h    = tid & 15;
    int n    = lane & 15;
    int k    = ((h & 8) << 1) + ((lane >> 4) << 3) + 32 * w + (h & 7);
    float v = 0.0f;
    if (n < 7 && k < 42) {
        int r = k / 7, c = k % 7;
        v = W[n * 111 + k];
        for (int j = 0; j < 69; ++j)
            v += W[n * 111 + 42 + j] * mask_val(j, r, c);
    }
    bop[tid] = (_Float16)v;
}

// out[B,7] = gs[B,42] @ Weff.T + b   via v_wmma_f32_16x16x32_f16.
// One 16-row tile per wave, grid-stride over tiles. Output staged through a
// per-wave LDS transpose so each tile is written as 28 coalesced b128 stores.
__global__ void __launch_bounds__(256) qgemm(const float* __restrict__ gs,
                                             const _Float16* __restrict__ bop,
                                             const float* __restrict__ bias,
                                             float* __restrict__ out,
                                             int nrows, int totalTiles, int totalWaves) {
    __shared__ __align__(16) float xpose[8][112];   // per-wave 16x7 staging

    const int lane      = threadIdx.x & 31;
    const int waveInBlk = threadIdx.x >> 5;
    const int gw        = blockIdx.x * (blockDim.x >> 5) + waveInBlk;
    const bool hi       = lane >= 16;
    const int  n        = lane & 15;       // output column this lane owns
    const int  k0       = hi ? 8 : 0;      // A-operand K base per ISA layout

    // B operands (uniform across waves; hot in WGP$/L2). 32B-aligned in d_ws.
    v16h b0 = *(const v16h*)(bop + lane * 16);
    v16h b1 = *(const v16h*)(bop + 512 + lane * 16);

    float bn = (n < 7) ? bias[n] : 0.0f;

    for (int t = gw; t < totalTiles; t += totalWaves) {
        // Prefetch next tile this wave will touch (tile = 2688 B; 32 lanes x 84 B).
        if (t + totalWaves < totalTiles) {
            const float* pf = gs + (size_t)(t + totalWaves) * 672 + lane * 21;
            __builtin_prefetch(pf, 0, 3);
        }

        int row = t * 16 + n;              // lane's A-matrix row (M = lane%16)
        if (row >= nrows) row = nrows - 1; // clamp (redundant loads, safe)
        const float* rp = gs + (size_t)row * 42;  // 8B aligned -> b64 loads

        v2f r0[4], r1[4], r2[4];
#pragma unroll
        for (int i = 0; i < 4; ++i) {
            r0[i] = *(const v2f*)(rp + k0 + 2 * i);        // K k0..k0+7
            r1[i] = *(const v2f*)(rp + k0 + 16 + 2 * i);   // K k0+16..k0+23
        }
        if (!hi) {
#pragma unroll
            for (int i = 0; i < 4; ++i)
                r2[i] = *(const v2f*)(rp + 32 + 2 * i);    // K 32..39 (in bounds)
        } else {
            r2[0] = *(const v2f*)(rp + 40);                // K 40..41 (in bounds)
            r2[1] = 0.0f; r2[2] = 0.0f; r2[3] = 0.0f;      // K 42..47 -> x0 in B
        }

        // Assemble f16 A operands per the 16-bit 16x32 A VGPR layout.
        v16h a0, a1;
#pragma unroll
        for (int i = 0; i < 4; ++i) {
            a0[2 * i]     = (_Float16)r0[i].x;
            a0[2 * i + 1] = (_Float16)r0[i].y;
            a0[8 + 2 * i] = (_Float16)r1[i].x;
            a0[9 + 2 * i] = (_Float16)r1[i].y;
            a1[2 * i]     = (_Float16)r2[i].x;
            a1[2 * i + 1] = (_Float16)r2[i].y;
            a1[8 + 2 * i] = (_Float16)0.0f;   // K >= 48: zero
            a1[9 + 2 * i] = (_Float16)0.0f;
        }

        // D = A0*B0 + A1*B1 (+0). EXEC is all ones here.
        v8f c = {};
        c = __builtin_amdgcn_wmma_f32_16x16x32_f16(false, a0, false, b0,
                                                   (short)0, c, false, false);
        c = __builtin_amdgcn_wmma_f32_16x16x32_f16(false, a1, false, b1,
                                                   (short)0, c, false, false);

        // C layout: lane owns column n; VGPR v holds row (hi?8:0)+v.
        int rowBase = t * 16;
        if (rowBase + 16 <= nrows) {
            // Fast path: stage 16x7 tile in LDS row-major, then 28 lanes
            // write 448 contiguous bytes as b128s (16B aligned: 448=28*16).
            if (n < 7) {
                int m0 = hi ? 8 : 0;
#pragma unroll
                for (int v = 0; v < 8; ++v)
                    xpose[waveInBlk][(m0 + v) * 7 + n] = c[v] + bn;
            }
            // Intra-wave LDS store->load: in-order per wave, no barrier needed.
            if (lane < 28) {
                float4 val = *(const float4*)(&xpose[waveInBlk][lane * 4]);
                *(float4*)(out + (size_t)t * 112 + lane * 4) = val;
            }
        } else {
            // Cold tail: per-element guarded scalar stores (int compares).
            if (n < 7) {
                int m0 = rowBase + (hi ? 8 : 0);
#pragma unroll
                for (int v = 0; v < 8; ++v) {
                    int rr = m0 + v;
                    if (rr < nrows) out[(size_t)rr * 7 + n] = c[v] + bn;
                }
            }
        }
    }
}

extern "C" void kernel_launch(void* const* d_in, const int* in_sizes, int n_in,
                              void* d_out, int out_size, void* d_ws, size_t ws_size,
                              hipStream_t stream) {
    const float* gs   = (const float*)d_in[0];   // [B,42] fp32
    const float* W    = (const float*)d_in[1];   // [7,111] fp32
    const float* bias = (const float*)d_in[2];   // [7] fp32
    float* out        = (float*)d_out;           // [B,7] fp32
    _Float16* bop     = (_Float16*)d_ws;         // 2 KB folded-weight operand

    int nrows = in_sizes[0] / 42;

    fold_weights<<<1, 1024, 0, stream>>>(W, bop);

    int totalTiles    = (nrows + 15) / 16;
    int wavesPerBlock = 256 / 32;
    int blocks        = 1024;                    // persistent-ish grid
    int totalWaves    = blocks * wavesPerBlock;
    if (totalWaves > totalTiles) {
        blocks     = (totalTiles + wavesPerBlock - 1) / wavesPerBlock;
        if (blocks < 1) blocks = 1;
        totalWaves = blocks * wavesPerBlock;
    }
    qgemm<<<blocks, 256, 0, stream>>>(gs, bop, bias, out,
                                      nrows, totalTiles, totalWaves);
}
